// SelfAttention_9921374454059
// MI455X (gfx1250) — matrix operand order repaired and checked by reading, not compiled, not run on hardware
//
#include <hip/hip_runtime.h>

typedef __attribute__((ext_vector_type(16))) _Float16 v16h;
typedef __attribute__((ext_vector_type(8)))  float    v8f;
typedef _Float16 h16;

#define BB      4
#define TT      1024
#define EE      1024
#define HH      16
#define DHD     64
#define KVLEN   2817
#define KVP     2848      /* 89*32, padded kv length */
#define MEMLEN  1793      /* KV - T */
#define POSROWS 3872      /* zero-padded pos table rows */

union FragU { v16h v; uint4 u[2]; };

__device__ __forceinline__ v16h ldfrag(const h16* p0, const h16* p1) {
  FragU t;
  t.u[0] = *(const uint4*)p0;
  t.u[1] = *(const uint4*)p1;
  return t.v;
}

__device__ __forceinline__ v8f wmma32(v16h a, v16h b, v8f c) {
  // D = A(16x32 f16) * B(32x16 f16) + C(16x16 f32)
  return __builtin_amdgcn_wmma_f32_16x16x32_f16(false, a, false, b, (short)0, c,
                                                false, false);
}

// 64x64 output tile per wave: 4 A-frags x 4 B-frags -> 16 WMMAs per K=32 step.
// A: rows [m .. m+63] of a row-major f16 matrix with stride EE (contiguous block)
// W: rows [n .. n+63] of row-major f16 weights [N][EE] (B = W^T)
__device__ __forceinline__ void gemm64x64(const h16* __restrict__ Abase,
                                          const h16* __restrict__ Wbase,
                                          v8f acc[16], int lane) {
  int ln = lane & 15, kh = lane >> 4;
  const h16* arow[4];
  const h16* wrow[4];
#pragma unroll
  for (int mt = 0; mt < 4; ++mt) arow[mt] = Abase + ((size_t)(mt * 16 + ln)) * EE;
#pragma unroll
  for (int nt = 0; nt < 4; ++nt) wrow[nt] = Wbase + ((size_t)(nt * 16 + ln)) * EE;

  for (int k0 = 0; k0 < EE; k0 += 32) {
    __builtin_prefetch(arow[0] + k0 + 256, 0, 1);
    __builtin_prefetch(wrow[0] + k0 + 256, 0, 1);
    v16h a[4];
#pragma unroll
    for (int mt = 0; mt < 4; ++mt)
      a[mt] = ldfrag(arow[mt] + k0 + 16 * kh, arow[mt] + k0 + 16 * kh + 8);
#pragma unroll
    for (int nt = 0; nt < 4; ++nt) {
      v16h bfr = ldfrag(wrow[nt] + k0 + 16 * kh, wrow[nt] + k0 + 16 * kh + 8);
#pragma unroll
      for (int mt = 0; mt < 4; ++mt)
        acc[nt * 4 + mt] = wmma32(a[mt], bfr, acc[nt * 4 + mt]);
    }
  }
}

// ---------------------------------------------------------------- conversions
__global__ void k_f32_to_f16(const float* __restrict__ s, h16* __restrict__ d, long n) {
  long stride = (long)gridDim.x * blockDim.x;
  for (long i = (long)blockIdx.x * blockDim.x + threadIdx.x; i < n; i += stride)
    d[i] = (h16)s[i];
}

// kv_input = concat(ltmem, cmem0, cmem1, mem, x) -> f16 [B][KVP][E], pad rows = 0
__global__ void k_build_kvin(const float* __restrict__ x, const float* __restrict__ mem,
                             const float* __restrict__ c0, const float* __restrict__ c1,
                             const float* __restrict__ lt, h16* __restrict__ kvin) {
  long n = (long)BB * KVP * EE;
  long stride = (long)gridDim.x * blockDim.x;
  for (long i = (long)blockIdx.x * blockDim.x + threadIdx.x; i < n; i += stride) {
    int e = (int)(i % EE);
    long re = i / EE;
    int r = (int)(re % KVP);
    int b = (int)(re / KVP);
    float v = 0.0f;
    if (r < 1)           v = lt [((long)b            )*EE + e];
    else if (r < 513)    v = c0 [((long)b*512 + (r-1  ))*EE + e];
    else if (r < 769)    v = c1 [((long)b*256 + (r-513))*EE + e];
    else if (r < 1793)   v = mem[((long)b*1024 + (r-769 ))*EE + e];
    else if (r < KVLEN)  v = x  [((long)b*1024 + (r-1793))*EE + e];
    kvin[i] = (h16)v;
  }
}

// pos table zero-padded to POSROWS so shifted index j + (T-1) - i is always in-bounds
__global__ void k_build_pos(const float* __restrict__ pos, h16* __restrict__ dst) {
  long n = (long)HH * POSROWS * DHD;
  long stride = (long)gridDim.x * blockDim.x;
  for (long i = (long)blockIdx.x * blockDim.x + threadIdx.x; i < n; i += stride) {
    int dh = (int)(i % DHD);
    long t = i / DHD;
    int jr = (int)(t % POSROWS);
    int h  = (int)(t / POSROWS);
    dst[i] = (jr < KVLEN) ? (h16)pos[((long)h*KVLEN + jr)*DHD + dh] : (h16)0.0f;
  }
}

// ---------------------------------------------------------------- projections
// q = x @ Wq^T, scaled by DH^-0.5, f16 out [B*T][E]. A rows come from kvin tail.
__global__ __launch_bounds__(128) void k_qproj(const h16* __restrict__ kvin,
                                               const h16* __restrict__ Wq,
                                               h16* __restrict__ qh) {
  int lane = threadIdx.x & 31, wave = threadIdx.x >> 5;
  int ln = lane & 15, kh = lane >> 4;
  int n0 = blockIdx.x * 256 + wave * 64;
  int m0 = blockIdx.y * 64;
  int b = m0 >> 10, t = m0 & 1023;          // 64 | T, so block stays in one b
  const h16* Abase = kvin + ((size_t)(b * KVP + MEMLEN + t)) * EE;
  v8f acc[16] = {};
  gemm64x64(Abase, Wq + (size_t)n0 * EE, acc, lane);
#pragma unroll
  for (int nt = 0; nt < 4; ++nt)
#pragma unroll
    for (int mt = 0; mt < 4; ++mt)
#pragma unroll
      for (int r = 0; r < 8; ++r)
        qh[((size_t)(m0 + mt*16 + r + 8*kh)) * EE + n0 + nt*16 + ln] =
            (h16)(acc[nt*4 + mt][r] * 0.125f);
}

// kv = kv_input @ Wkv^T -> k f16 [B][KVP][E], v transposed f16 [B][H][DH][KVP]
__global__ __launch_bounds__(128) void k_kvproj(const h16* __restrict__ kvin,
                                                const h16* __restrict__ Wkv,
                                                h16* __restrict__ kbuf,
                                                h16* __restrict__ vT) {
  int lane = threadIdx.x & 31, wave = threadIdx.x >> 5;
  int ln = lane & 15, kh = lane >> 4;
  int n0 = blockIdx.x * 256 + wave * 64;
  int m0 = blockIdx.y * 64;
  const h16* Abase = kvin + ((size_t)m0) * EE;   // pad rows are zero
  v8f acc[16] = {};
  gemm64x64(Abase, Wkv + (size_t)n0 * EE, acc, lane);
#pragma unroll
  for (int nt = 0; nt < 4; ++nt) {
    int nglob = n0 + nt*16 + ln;
    if (nglob < EE) {
#pragma unroll
      for (int mt = 0; mt < 4; ++mt)
#pragma unroll
        for (int r = 0; r < 8; ++r) {
          int row = m0 + mt*16 + r + 8*kh;
          int b = row / KVP, kv = row % KVP;
          kbuf[((size_t)b*KVP + kv) * EE + nglob] = (h16)acc[nt*4 + mt][r];
        }
    } else {
      int h = (nglob - EE) >> 6, dh = (nglob - EE) & 63;
#pragma unroll
      for (int mt = 0; mt < 4; ++mt)
#pragma unroll
        for (int r = 0; r < 8; ++r) {
          int row = m0 + mt*16 + r + 8*kh;
          int b = row / KVP, kv = row % KVP;
          vT[(((size_t)b*HH + h)*DHD + dh) * KVP + kv] = (h16)acc[nt*4 + mt][r];
        }
    }
  }
}

// logits = attn_out @ Wout^T + bout, f32 output
__global__ __launch_bounds__(128) void k_outproj(const h16* __restrict__ A,
                                                 const h16* __restrict__ Wout,
                                                 const float* __restrict__ bout,
                                                 float* __restrict__ out) {
  int lane = threadIdx.x & 31, wave = threadIdx.x >> 5;
  int ln = lane & 15, kh = lane >> 4;
  int n0 = blockIdx.x * 256 + wave * 64;
  int m0 = blockIdx.y * 64;
  v8f acc[16] = {};
  gemm64x64(A + (size_t)m0 * EE, Wout + (size_t)n0 * EE, acc, lane);
#pragma unroll
  for (int nt = 0; nt < 4; ++nt) {
    float bias = bout[n0 + nt*16 + ln];
#pragma unroll
    for (int mt = 0; mt < 4; ++mt)
#pragma unroll
      for (int r = 0; r < 8; ++r)
        out[((size_t)(m0 + mt*16 + r + 8*kh)) * EE + n0 + nt*16 + ln] =
            acc[nt*4 + mt][r] + bias;
  }
}

// ---------------------------------------------------------------- attention
// one wave per (b, h, 16-query tile); streaming online softmax over 32 kv/iter
__global__ __launch_bounds__(32) void k_attn(const h16* __restrict__ qh,
                                             const h16* __restrict__ kbuf,
                                             const h16* __restrict__ vT,
                                             const h16* __restrict__ posPad,
                                             h16* __restrict__ attnOut) {
  __shared__ float sPos[16][48];
  __shared__ float sS[16][32];
  __shared__ h16   sP[16][32];
  __shared__ float sScale[16];

  int lane = threadIdx.x;
  int ln = lane & 15, kh = lane >> 4;
  int blk = blockIdx.x;
  int qt = blk & 63;                  // T/16 = 64
  int h  = (blk >> 6) & 15;
  int b  = blk >> 10;
  int i0 = qt * 16;

  // Q fragments (scale already folded into qh)
  const h16* qbase = qh + ((size_t)(b*TT + i0 + ln)) * EE + h*DHD;
  v16h qf0 = ldfrag(qbase      + 16*kh, qbase      + 16*kh + 8);
  v16h qf1 = ldfrag(qbase + 32 + 16*kh, qbase + 32 + 16*kh + 8);

  v8f o[4] = {};
  float m_run = -3.0e38f, l_run = 0.0f;

  int jEnd = (i0 + 16 + MEMLEN + 31) & ~31;   // exclusive, rounded to 32
  if (jEnd > KVP) jEnd = KVP;
  int jjb = 1008 - i0;                        // band start offset (j0 + T-1 - (i0+15))

  for (int j0 = 0; j0 < jEnd; j0 += 32) {
    // ---- S = Q K^T (two 16-col tiles)
#pragma unroll
    for (int st = 0; st < 2; ++st) {
      const h16* kb = kbuf + ((size_t)(b*KVP + j0 + st*16 + ln)) * EE + h*DHD;
      v8f s = {};
      s = wmma32(qf0, ldfrag(kb      + 16*kh, kb      + 16*kh + 8), s);
      s = wmma32(qf1, ldfrag(kb + 32 + 16*kh, kb + 32 + 16*kh + 8), s);
#pragma unroll
      for (int r = 0; r < 8; ++r) sS[r + 8*kh][st*16 + ln] = s[r];
    }
    // ---- pos band: rows jjb+j0 .. +47 cover the 31-diag shift of both S tiles
#pragma unroll
    for (int ct = 0; ct < 3; ++ct) {
      const h16* pb = posPad + ((size_t)h*POSROWS + (j0 + jjb) + ct*16 + ln) * DHD;
      v8f p = {};
      p = wmma32(qf0, ldfrag(pb      + 16*kh, pb      + 16*kh + 8), p);
      p = wmma32(qf1, ldfrag(pb + 32 + 16*kh, pb + 32 + 16*kh + 8), p);
#pragma unroll
      for (int r = 0; r < 8; ++r) sPos[r + 8*kh][ct*16 + ln] = p[r];
    }
    __syncthreads();

    // ---- online softmax; lane t owns query row t
    if (lane < 16) {
      int t = lane, iq = i0 + t;
      float vals[32], rmax = -3.0e38f;
#pragma unroll
      for (int jj = 0; jj < 32; ++jj) {
        int j = j0 + jj;
        float s = sS[t][jj] + sPos[t][jj + 15 - t];   // shift trick
        bool masked = (j >= KVLEN) || (j > iq + MEMLEN);
        s = masked ? -3.0e38f : s;
        vals[jj] = s;
        rmax = fmaxf(rmax, s);
      }
      float mnew = fmaxf(m_run, rmax);
      float sc = __expf(m_run - mnew);
      float ss = 0.0f;
#pragma unroll
      for (int jj = 0; jj < 32; ++jj) {
        float p = __expf(vals[jj] - mnew);
        ss += p;
        sP[t][jj] = (h16)p;
      }
      l_run = l_run * sc + ss;
      m_run = mnew;
      sScale[t] = sc;
    }
    __syncthreads();

    // ---- rescale O, then O += P V  (P: 16x32, V: 32x64)
    float scl[8];
#pragma unroll
    for (int r = 0; r < 8; ++r) scl[r] = sScale[r + 8*kh];
    v16h pf = ldfrag(&sP[ln][16*kh], &sP[ln][16*kh + 8]);
#pragma unroll
    for (int nt = 0; nt < 4; ++nt) {
#pragma unroll
      for (int r = 0; r < 8; ++r) o[nt][r] *= scl[r];
      const h16* vb = vT + (((size_t)(b*HH + h))*DHD + nt*16 + ln) * KVP + j0 + 16*kh;
      o[nt] = wmma32(pf, ldfrag(vb, vb + 8), o[nt]);
    }
    __syncthreads();
  }

  if (lane < 16) sScale[lane] = (l_run > 0.0f) ? (1.0f / l_run) : 0.0f;
  __syncthreads();
  float inv[8];
#pragma unroll
  for (int r = 0; r < 8; ++r) inv[r] = sScale[r + 8*kh];
#pragma unroll
  for (int nt = 0; nt < 4; ++nt)
#pragma unroll
    for (int r = 0; r < 8; ++r)
      attnOut[((size_t)(b*TT + i0 + r + 8*kh)) * EE + h*DHD + nt*16 + ln] =
          (h16)(o[nt][r] * inv[r]);
}

// ---------------------------------------------------------------- launcher
extern "C" void kernel_launch(void* const* d_in, const int* in_sizes, int n_in,
                              void* d_out, int out_size, void* d_ws, size_t ws_size,
                              hipStream_t stream) {
  const float* x    = (const float*)d_in[0];
  const float* mem  = (const float*)d_in[1];
  const float* c0   = (const float*)d_in[2];
  const float* c1   = (const float*)d_in[3];
  const float* lt   = (const float*)d_in[4];
  const float* pos  = (const float*)d_in[5];
  const float* Wq   = (const float*)d_in[6];
  const float* Wkv  = (const float*)d_in[7];
  const float* Wout = (const float*)d_in[8];
  const float* bout = (const float*)d_in[9];
  float* out = (float*)d_out;

  char* ws = (char*)d_ws;
  size_t off = 0;
  auto alloc = [&](size_t bytes) -> char* {
    char* p = ws + off;
    off = (off + bytes + 255) & ~(size_t)255;
    return p;
  };
  h16* kvin   = (h16*)alloc((size_t)BB * KVP * EE * 2);
  h16* Wq_h   = (h16*)alloc((size_t)EE * EE * 2);
  h16* Wkv_h  = (h16*)alloc((size_t)2 * EE * EE * 2);
  h16* Wout_h = (h16*)alloc((size_t)EE * EE * 2);
  h16* qh     = (h16*)alloc((size_t)BB * TT * EE * 2);
  h16* kbuf   = (h16*)alloc((size_t)BB * KVP * EE * 2);
  h16* vT     = (h16*)alloc((size_t)BB * HH * DHD * KVP * 2);
  h16* posPad = (h16*)alloc((size_t)HH * POSROWS * DHD * 2);
  h16* aOut   = (h16*)alloc((size_t)BB * TT * EE * 2);

  k_build_kvin<<<8192, 256, 0, stream>>>(x, mem, c0, c1, lt, kvin);
  k_f32_to_f16<<<2048, 256, 0, stream>>>(Wq,   Wq_h,   (long)EE*EE);
  k_f32_to_f16<<<4096, 256, 0, stream>>>(Wkv,  Wkv_h,  (long)2*EE*EE);
  k_f32_to_f16<<<2048, 256, 0, stream>>>(Wout, Wout_h, (long)EE*EE);
  k_build_pos <<<4096, 256, 0, stream>>>(pos, posPad);

  k_qproj  <<<dim3(EE/256, (BB*TT)/64),    128, 0, stream>>>(kvin, Wq_h, qh);
  k_kvproj <<<dim3(2*EE/256, (BB*KVP)/64), 128, 0, stream>>>(kvin, Wkv_h, kbuf, vT);
  k_attn   <<<dim3(BB*HH*(TT/16)),         32,  0, stream>>>(qh, kbuf, vT, posPad, aOut);
  k_outproj<<<dim3(EE/256, (BB*TT)/64),    128, 0, stream>>>(aOut, Wout_h, bout, out);
}
